// LMAR_model_1520418422926
// MI455X (gfx1250) — matrix operand order repaired and checked by reading, not compiled, not run on hardware
//
#include <hip/hip_runtime.h>
#include <hip/hip_bf16.h>

typedef __attribute__((ext_vector_type(16))) _Float16 v16h;
typedef __attribute__((ext_vector_type(8)))  _Float16 v8h;
typedef __attribute__((ext_vector_type(8)))  float    v8f;

#define B_   2
#define H_   512
#define W_   512
#define DH_  256
#define DW_  256
#define NPIX (B_*H_*W_)

// ---------------- bicubic helper (Keys, a = -0.5) ----------------
__device__ __forceinline__ float cubw(float t) {
  t = fabsf(t);
  float t2 = t*t, t3 = t2*t;
  if (t <= 1.f) return 1.5f*t3 - 2.5f*t2 + 1.f;
  if (t <  2.f) return -0.5f*t3 + 2.5f*t2 - 4.f*t + 2.f;
  return 0.f;
}

// Generic separable bicubic resize (jax.image.resize, antialias=False).
// in: [C2, ih, iw] -> out: [C2, oh, ow]; edge weights renormalized per-dim.
__global__ void resize_bicubic(const float* __restrict__ in, float* __restrict__ out,
                               int C2, int ih, int iw, int oh, int ow) {
  int idx = blockIdx.x * blockDim.x + threadIdx.x;
  int total = C2 * oh * ow;
  if (idx >= total) return;
  int ox = idx % ow;
  int t  = idx / ow;
  int oy = t % oh;
  int ch = t / oh;
  float sy = (oy + 0.5f) * (float)ih / (float)oh - 0.5f;
  float sx = (ox + 0.5f) * (float)iw / (float)ow - 0.5f;
  int yb = (int)floorf(sy) - 1;
  int xb = (int)floorf(sx) - 1;
  float wy[4], wx[4], sumy = 0.f, sumx = 0.f;
  #pragma unroll
  for (int d = 0; d < 4; ++d) {
    int yy = yb + d;
    wy[d] = (yy >= 0 && yy < ih) ? cubw(sy - (float)yy) : 0.f;
    sumy += wy[d];
    int xx = xb + d;
    wx[d] = (xx >= 0 && xx < iw) ? cubw(sx - (float)xx) : 0.f;
    sumx += wx[d];
  }
  const float* src = in + (size_t)ch * ih * iw;
  float acc = 0.f;
  #pragma unroll
  for (int dy = 0; dy < 4; ++dy) {
    int yy = min(max(yb + dy, 0), ih - 1);
    float a = 0.f;
    #pragma unroll
    for (int dx = 0; dx < 4; ++dx) {
      int xx = min(max(xb + dx, 0), iw - 1);
      a += wx[dx] * src[(size_t)yy * iw + xx];
    }
    acc += wy[dy] * a;
  }
  out[idx] = acc / (sumy * sumx);
}

// Fused: bicubic upsample down_x 256->512, laplacian = x - up, rel coords,
// emit padded 8-half f16 MLP input rows [rely, relx, 1, 1, lap0..2, 0].
__global__ void build_input(const float* __restrict__ xg, const float* __restrict__ dxg,
                            _Float16* __restrict__ inp8) {
  int idx = blockIdx.x * blockDim.x + threadIdx.x;
  if (idx >= NPIX) return;
  int j = idx % W_;
  int t = idx / W_;
  int i = t % H_;
  int b = t / H_;

  float sy = (i + 0.5f) * 0.5f - 0.5f;
  float sx = (j + 0.5f) * 0.5f - 0.5f;
  int yb = (int)floorf(sy) - 1;
  int xb = (int)floorf(sx) - 1;
  float wy[4], wx[4], sumy = 0.f, sumx = 0.f;
  #pragma unroll
  for (int d = 0; d < 4; ++d) {
    int yy = yb + d;
    wy[d] = (yy >= 0 && yy < DH_) ? cubw(sy - (float)yy) : 0.f;
    sumy += wy[d];
    int xx = xb + d;
    wx[d] = (xx >= 0 && xx < DW_) ? cubw(sx - (float)xx) : 0.f;
    sumx += wx[d];
  }
  float inv = 1.f / (sumy * sumx);

  float lap[3];
  #pragma unroll
  for (int c = 0; c < 3; ++c) {
    const float* src = dxg + ((size_t)(b * 3 + c)) * DH_ * DW_;
    float acc = 0.f;
    #pragma unroll
    for (int dy = 0; dy < 4; ++dy) {
      int yy = min(max(yb + dy, 0), DH_ - 1);
      float a = 0.f;
      #pragma unroll
      for (int dx = 0; dx < 4; ++dx) {
        int xx = min(max(xb + dx, 0), DW_ - 1);
        a += wx[dx] * src[(size_t)yy * DW_ + xx];
      }
      acc += wy[dy] * a;
    }
    float up = acc * inv;
    lap[c] = xg[((size_t)(b * 3 + c) * H_ + i) * W_ + j] - up;
  }

  float gy = -1.f + 1.f / H_ + (2.f / H_) * (float)i;
  float gx = -1.f + 1.f / W_ + (2.f / W_) * (float)j;
  gy = fminf(fmaxf(gy, -1.f + 1e-6f), 1.f - 1e-6f);
  gx = fminf(fmaxf(gx, -1.f + 1e-6f), 1.f - 1e-6f);
  float fy = rintf(((gy + 1.f) * DH_ - 1.f) * 0.5f);
  float fx = rintf(((gx + 1.f) * DW_ - 1.f) * 0.5f);
  int iy = (int)fminf(fmaxf(fy, 0.f), (float)(DH_ - 1));
  int ix = (int)fminf(fmaxf(fx, 0.f), (float)(DW_ - 1));
  float qy = -1.f + 1.f / DH_ + (2.f / DH_) * (float)iy;
  float qx = -1.f + 1.f / DW_ + (2.f / DW_) * (float)ix;
  float rely = (gy - qy) * (float)DH_;
  float relx = (gx - qx) * (float)DW_;

  _Float16* o = inp8 + (size_t)idx * 8;
  o[0] = (_Float16)rely; o[1] = (_Float16)relx;
  o[2] = (_Float16)1.f;  o[3] = (_Float16)1.f;
  o[4] = (_Float16)lap[0]; o[5] = (_Float16)lap[1]; o[6] = (_Float16)lap[2];
  o[7] = (_Float16)0.f;
}

// Pack an f32 weight matrix [Ksrc x Nsrc] into B-fragment-contiguous f16:
// fragment f = s*NT + t; per lane l, element e: k = 32s + (e<8 ? 8*(l>>4)+e : 16+8*(l>>4)+e-8),
// n = 16t + (l&15). One (frag,lane) per thread.
__global__ void prepack(const float* __restrict__ src, _Float16* __restrict__ dst,
                        int Ksrc, int Nsrc, int NT, int nfrag) {
  int idx = blockIdx.x * blockDim.x + threadIdx.x;
  if (idx >= nfrag * 32) return;
  int f = idx >> 5, l = idx & 31;
  int s = f / NT, t = f % NT;
  int n = t * 16 + (l & 15);
  _Float16* d = dst + ((size_t)f * 32 + l) * 16;
  #pragma unroll
  for (int e = 0; e < 16; ++e) {
    int k = s * 32 + (e < 8 ? (l >> 4) * 8 + e : 16 + (l >> 4) * 8 + (e - 8));
    float v = (k < Ksrc && n < Nsrc) ? src[(size_t)k * Nsrc + n] : 0.f;
    d[e] = (_Float16)v;
  }
}

// ---------------- fused MLP (6 layers, WMMA) + einsum ----------------
// One wave owns a 16-pixel tile. Activations live in a wave-private LDS tile.
__global__ __launch_bounds__(256) void mlp_einsum(
    const _Float16* __restrict__ inp8, const _Float16* __restrict__ wpack,
    const float* __restrict__ xg, float* __restrict__ out_raw) {
  __shared__ __align__(16) unsigned char smem[8 * 16 * 96 * 4];  // 48 KB
  int lane = threadIdx.x & 31;
  int wave = threadIdx.x >> 5;
  int p0 = (blockIdx.x * 8 + wave) * 16;
  _Float16* act = (_Float16*)(smem + (size_t)wave * (16 * 96 * 4));
  float*    fin = (float*)   (smem + (size_t)wave * (16 * 96 * 4));
  int m = lane & 15, g = lane >> 4;

  // ---- layer 0: K padded 7->32 (inp8 rows already zero-padded to 8) ----
  v16h A0 = {};
  if (g == 0) {
    v8h lo = *(const v8h*)(inp8 + (size_t)(p0 + m) * 8);
    #pragma unroll
    for (int e = 0; e < 8; ++e) A0[e] = lo[e];
  }
  #pragma unroll
  for (int t = 0; t < 8; ++t) {
    v16h bfr = *(const v16h*)(wpack + ((size_t)t * 32 + lane) * 16);
    v8f acc = {};
    acc = __builtin_amdgcn_wmma_f32_16x16x32_f16(false, A0, false, bfr,
                                                 (short)0, acc, false, false);
    #pragma unroll
    for (int r = 0; r < 8; ++r) {
      float v = acc[r];
      v = v > 0.f ? v : 0.01f * v;                   // leaky_relu(0.01)
      act[(r + 8 * g) * 128 + t * 16 + m] = (_Float16)v;
    }
  }

  // ---- hidden layers 1..4: 128x128 ----
  #pragma unroll
  for (int layer = 0; layer < 4; ++layer) {
    int base = 8 + layer * 32;
    v16h A[4];
    #pragma unroll
    for (int s = 0; s < 4; ++s) {   // A layout: e<8 -> K=8g+e ; e>=8 -> K=16+8g+(e-8)
      const _Float16* ap = act + m * 128 + s * 32 + 8 * g;
      v8h lo = *(const v8h*)ap;
      v8h hi = *(const v8h*)(ap + 16);
      #pragma unroll
      for (int e = 0; e < 8; ++e) { A[s][e] = lo[e]; A[s][e + 8] = hi[e]; }
    }
    #pragma unroll
    for (int t = 0; t < 8; ++t) {
      v8f acc = {};
      #pragma unroll
      for (int s = 0; s < 4; ++s) {
        v16h bfr = *(const v16h*)(wpack + ((size_t)(base + s * 8 + t) * 32 + lane) * 16);
        acc = __builtin_amdgcn_wmma_f32_16x16x32_f16(false, A[s], false, bfr,
                                                     (short)0, acc, false, false);
      }
      #pragma unroll
      for (int r = 0; r < 8; ++r) {
        float v = acc[r];
        v = v > 0.f ? v : 0.01f * v;
        act[(r + 8 * g) * 128 + t * 16 + m] = (_Float16)v;
      }
    }
  }

  // ---- output layer: 128 -> 96 (81 padded), f32 to LDS, no relu ----
  {
    v16h A[4];
    #pragma unroll
    for (int s = 0; s < 4; ++s) {
      const _Float16* ap = act + m * 128 + s * 32 + 8 * g;
      v8h lo = *(const v8h*)ap;
      v8h hi = *(const v8h*)(ap + 16);
      #pragma unroll
      for (int e = 0; e < 8; ++e) { A[s][e] = lo[e]; A[s][e + 8] = hi[e]; }
    }
    #pragma unroll
    for (int t = 0; t < 6; ++t) {
      v8f acc = {};
      #pragma unroll
      for (int s = 0; s < 4; ++s) {
        v16h bfr = *(const v16h*)(wpack + ((size_t)(136 + s * 6 + t) * 32 + lane) * 16);
        acc = __builtin_amdgcn_wmma_f32_16x16x32_f16(false, A[s], false, bfr,
                                                     (short)0, acc, false, false);
      }
      #pragma unroll
      for (int r = 0; r < 8; ++r)
        fin[(r + 8 * g) * 96 + t * 16 + m] = acc[r];
    }
  }

  // ---- einsum: out[c] = sum_k unfold3(x)[k] * lw[k][c], k = c_in*9+ki*3+kj ----
  if (lane < 16) {
    int P = p0 + lane;
    int b = P / (H_ * W_);
    int rem = P - b * (H_ * W_);
    int i = rem / W_;
    int j = rem - i * W_;
    const float* lw = fin + lane * 96;
    float o0 = 0.f, o1 = 0.f, o2 = 0.f;
    #pragma unroll
    for (int c = 0; c < 3; ++c) {
      const float* xc = xg + (size_t)(b * 3 + c) * H_ * W_;
      #pragma unroll
      for (int ki = 0; ki < 3; ++ki) {
        int y = i + ki - 1;
        #pragma unroll
        for (int kj = 0; kj < 3; ++kj) {
          int xx = j + kj - 1;
          float v = (y >= 0 && y < H_ && xx >= 0 && xx < W_)
                        ? xc[(size_t)y * W_ + xx] : 0.f;
          int kk = (c * 9 + ki * 3 + kj) * 3;
          o0 += v * lw[kk]; o1 += v * lw[kk + 1]; o2 += v * lw[kk + 2];
        }
      }
    }
    size_t o = (size_t)(b * 3) * H_ * W_ + (size_t)i * W_ + j;
    out_raw[o]                       = o0;
    out_raw[o + (size_t)H_ * W_]     = o1;
    out_raw[o + 2 * (size_t)H_ * W_] = o2;
  }
}

// ---------------- 3x3 stride-2 pad-1 conv ----------------
__global__ void conv3x3s2(const float* __restrict__ in, const float* __restrict__ wgt,
                          const float* __restrict__ bias, float* __restrict__ out,
                          int Bn, int Cin, int H, int W, int Cout) {
  int OH = H / 2, OW = W / 2;
  long long idx = (long long)blockIdx.x * blockDim.x + threadIdx.x;
  long long total = (long long)Bn * Cout * OH * OW;
  if (idx >= total) return;
  int ox = (int)(idx % OW); long long t = idx / OW;
  int oy = (int)(t % OH); t /= OH;
  int co = (int)(t % Cout); int b = (int)(t / Cout);
  float acc = bias[co];
  for (int ci = 0; ci < Cin; ++ci) {
    const float* src = in + (size_t)(b * Cin + ci) * H * W;
    const float* wp  = wgt + (size_t)(co * Cin + ci) * 9;
    #pragma unroll
    for (int ki = 0; ki < 3; ++ki) {
      int y = 2 * oy + ki - 1;
      if (y < 0 || y >= H) continue;
      #pragma unroll
      for (int kj = 0; kj < 3; ++kj) {
        int x = 2 * ox + kj - 1;
        if (x < 0 || x >= W) continue;
        acc += src[(size_t)y * W + x] * wp[ki * 3 + kj];
      }
    }
  }
  out[idx] = acc;
}

// ---------------- 1x1 conv over concat(down_x, out): 6 -> 3 ----------------
__global__ void mod1x1(const float* __restrict__ dxp, const float* __restrict__ outp,
                       const float* __restrict__ w, const float* __restrict__ bias,
                       float* __restrict__ dst, int Bn, int HW) {
  int idx = blockIdx.x * blockDim.x + threadIdx.x;
  int total = Bn * 3 * HW;
  if (idx >= total) return;
  int p = idx % HW; int t = idx / HW;
  int co = t % 3; int b = t / 3;
  float acc = bias[co];
  #pragma unroll
  for (int ci = 0; ci < 3; ++ci)
    acc += w[co * 6 + ci] * dxp[(size_t)(b * 3 + ci) * HW + p];
  #pragma unroll
  for (int ci = 3; ci < 6; ++ci)
    acc += w[co * 6 + ci] * outp[(size_t)(b * 3 + (ci - 3)) * HW + p];
  dst[idx] = acc;
}

extern "C" void kernel_launch(void* const* d_in, const int* in_sizes, int n_in,
                              void* d_out, int out_size, void* d_ws, size_t ws_size,
                              hipStream_t stream) {
  (void)in_sizes; (void)n_in; (void)out_size; (void)ws_size;
  const float* x      = (const float*)d_in[0];
  const float* W_in   = (const float*)d_in[1];
  const float* W_h1   = (const float*)d_in[2];
  const float* W_h2   = (const float*)d_in[3];
  const float* W_h3   = (const float*)d_in[4];
  const float* W_h4   = (const float*)d_in[5];
  const float* W_out  = (const float*)d_in[6];
  const float* feat_w = (const float*)d_in[7];
  const float* feat_b = (const float*)d_in[8];
  const float* mod_w  = (const float*)d_in[9];
  const float* mod_b  = (const float*)d_in[10];

  // workspace layout (bytes)
  char* ws = (char*)d_ws;
  _Float16* wpack   = (_Float16*)ws;                 //   163,840 B (160 frags * 512 halves)
  float*    down_x  = (float*)(ws + 163840);         // 1,572,864 B (2,3,256,256)
  _Float16* inp8    = (_Float16*)(ws + 1736704);     // 8,388,608 B (NPIX * 8 halves)
  float*    out_raw = (float*)(ws + 10125312);       // 6,291,456 B (2,3,512,512)
  float*    lrtmp   = (float*)(ws + 16416768);       // 8,388,608 B (2,64,128,128)

  // output regions (flat, return order)
  float* o_mod = (float*)d_out;          // down_x_mod (2,3,256,256)
  float* o_hr  = o_mod + 393216;         // hr_feature (2,64,256,256)
  float* o_new = o_hr  + 8388608;        // new_lr_feature
  float* o_ori = o_new + 8388608;        // ori_lr_feature
  float* o_out = o_ori + 8388608;        // out (2,3,256,256)

  const int thr = 256;

  // 1) pack MLP weights into f16 B-fragments
  prepack<<<1, thr, 0, stream>>>(W_in,  wpack,               7, 128, 8,  8);
  prepack<<<4, thr, 0, stream>>>(W_h1,  wpack +   8 * 512, 128, 128, 8, 32);
  prepack<<<4, thr, 0, stream>>>(W_h2,  wpack +  40 * 512, 128, 128, 8, 32);
  prepack<<<4, thr, 0, stream>>>(W_h3,  wpack +  72 * 512, 128, 128, 8, 32);
  prepack<<<4, thr, 0, stream>>>(W_h4,  wpack + 104 * 512, 128, 128, 8, 32);
  prepack<<<3, thr, 0, stream>>>(W_out, wpack + 136 * 512, 128,  81, 6, 24);

  // 2) down_x = bicubic(x, 512->256)
  resize_bicubic<<<(6 * 256 * 256 + thr - 1) / thr, thr, 0, stream>>>(
      x, down_x, 6, 512, 512, 256, 256);

  // 3) laplacian + rel coords -> padded f16 MLP inputs
  build_input<<<(NPIX + thr - 1) / thr, thr, 0, stream>>>(x, down_x, inp8);

  // 4) fused 6-layer WMMA MLP + 27x3 einsum -> out_raw (2,3,512,512)
  mlp_einsum<<<NPIX / 128, 256, 0, stream>>>(inp8, wpack, x, out_raw);

  // 5) out = bicubic(out_raw, 512->256)  (final output #5)
  resize_bicubic<<<(6 * 256 * 256 + thr - 1) / thr, thr, 0, stream>>>(
      out_raw, o_out, 6, 512, 512, 256, 256);

  // 6) down_x_mod = 1x1 conv over concat(down_x, out)  (output #1)
  mod1x1<<<(2 * 3 * 65536 + thr - 1) / thr, thr, 0, stream>>>(
      down_x, o_out, mod_w, mod_b, o_mod, 2, 65536);

  // 7) hr_feature = conv(x) stride2  (output #2)
  conv3x3s2<<<(2 * 64 * 256 * 256 + thr - 1) / thr, thr, 0, stream>>>(
      x, feat_w, feat_b, o_hr, 2, 3, 512, 512, 64);

  // 8) ori_lr_feature = resize(conv(down_x), 128->256)  (output #4)
  conv3x3s2<<<(2 * 64 * 128 * 128 + thr - 1) / thr, thr, 0, stream>>>(
      down_x, feat_w, feat_b, lrtmp, 2, 3, 256, 256, 64);
  resize_bicubic<<<(128 * 256 * 256 + thr - 1) / thr, thr, 0, stream>>>(
      lrtmp, o_ori, 128, 128, 128, 256, 256);

  // 9) new_lr_feature = resize(conv(down_x_mod), 128->256)  (output #3)
  conv3x3s2<<<(2 * 64 * 128 * 128 + thr - 1) / thr, thr, 0, stream>>>(
      o_mod, feat_w, feat_b, lrtmp, 2, 3, 256, 256, 64);
  resize_bicubic<<<(128 * 256 * 256 + thr - 1) / thr, thr, 0, stream>>>(
      lrtmp, o_new, 128, 128, 128, 256, 256);
}
